// AttentionLayer_8555574854347
// MI455X (gfx1250) — compile-verified, matrix operand order checked by reading, not run on hardware
//
#include <hip/hip_runtime.h>

typedef _Float16 f16;
typedef __attribute__((ext_vector_type(16))) _Float16 v16h;
typedef __attribute__((ext_vector_type(8)))  _Float16 v8h;
typedef __attribute__((ext_vector_type(8)))  float    v8f;
typedef __attribute__((ext_vector_type(2)))  float    v2f;
typedef __attribute__((ext_vector_type(4)))  unsigned int v4u;
typedef __attribute__((ext_vector_type(8)))  int      v8i;
typedef __attribute__((ext_vector_type(4)))  int      v4i;

#define BB   4
#define CC   256
#define HH   64
#define WW   64
#define NN   4096
#define HP   66
#define NP   (HP * HP)      // padded spatial (zero halo) -> branch-free im2col
#define DD   64
#define KTOP 819
#define KPAD 832
#define K9C  (9 * CC)

// ---- WMMA fragment loaders -------------------------------------------------
// f16 A/B fragment for 16x16x32: lane l covers row (l&15); the 16 halfs are
// K = (l>>4)*8 + [0..7] and K = (l>>4)*8 + 16 + [0..7]  -> two b128 loads.
__device__ __forceinline__ v16h ld_frag16(const f16* p) {
  v8h lo = *reinterpret_cast<const v8h*>(p);
  v8h hi = *reinterpret_cast<const v8h*>(p + 16);
  v16h r;
#pragma unroll
  for (int i = 0; i < 8; ++i) { r[i] = lo[i]; r[i + 8] = hi[i]; }
  return r;
}
// f32 A/B fragment for 16x16x4: lane l covers row (l&15), K = (l>>4)*2 + [0,1]
__device__ __forceinline__ v2f ld2f(const float* p) {
  return *reinterpret_cast<const v2f*>(p);
}
__device__ __forceinline__ v8f wmma16(v16h a, v16h b, v8f c) {
  return __builtin_amdgcn_wmma_f32_16x16x32_f16(false, a, false, b, (short)0, c, false, false);
}
__device__ __forceinline__ v8f wmma4(v2f a, v2f b, v8f c) {
  return __builtin_amdgcn_wmma_f32_16x16x4_f32(false, a, false, b, (short)0, c, false, false);
}

// ---- weight prep -----------------------------------------------------------
// Fold BN into 3x3 conv weights; repack [co][ci][ky][kx] -> [co][tap*C + ci] fp16
__global__ void k_fold3(const float* __restrict__ w, const float* __restrict__ sc,
                        const float* __restrict__ bi, const float* __restrict__ mu,
                        const float* __restrict__ va, f16* __restrict__ w16,
                        float* __restrict__ bvec) {
  int g = blockIdx.x * 256 + threadIdx.x;
  if (g >= CC * CC * 9) return;
  int co = g / (CC * 9);
  int r  = g - co * (CC * 9);
  int ci = r / 9;
  int tap = r - ci * 9;
  float inv = sc[co] / sqrtf(va[co] + 1e-5f);
  w16[(size_t)co * K9C + tap * CC + ci] = (f16)(w[g] * inv);
  if (r == 0) bvec[co] = bi[co] - mu[co] * inv;
}

__global__ void k_pack1(const float* __restrict__ w, f16* __restrict__ w16, int n) {
  int g = blockIdx.x * 256 + threadIdx.x;
  if (g < n) w16[g] = (f16)w[g];
}

// xinP[b][np][c] = fp16(x[b][c][n]) with 1-pixel zero halo (66x66)
__global__ void k_tin_pad(const float* __restrict__ x, f16* __restrict__ xT) {
  int g = blockIdx.x * 256 + threadIdx.x;
  if (g >= BB * NP * CC) return;
  int b = g / (NP * CC);
  int r = g - b * (NP * CC);
  int np = r / CC;
  int c  = r - np * CC;
  int hh = np / HP, ww = np - hh * HP;
  int h = hh - 1, w = ww - 1;
  f16 v = (f16)0.f;
  if (h >= 0 && h < HH && w >= 0 && w < WW)
    v = (f16)x[((size_t)b * CC + c) * NN + (h << 6) + w];
  xT[g] = v;
}

// ---- 3x3 conv as implicit GEMM with f16 WMMA (padded input, branch-free) ---
__global__ void __launch_bounds__(256)
k_conv3(const f16* __restrict__ inP, const f16* __restrict__ w,
        const float* __restrict__ bias, const float* __restrict__ pos,
        float* __restrict__ lin32, f16* __restrict__ lin16,
        f16* __restrict__ outT16) {
  int wid  = (blockIdx.x * 256 + threadIdx.x) >> 5;
  int lane = threadIdx.x & 31;
  if (wid >= BB * (NN / 16) * (CC / 16)) return;
  int ct = wid & 15;
  int nt = (wid >> 4) & 255;
  int b  = wid >> 12;
  int col15 = lane & 15;
  int khalf = (lane >> 4) << 3;
  int co = (ct << 4) + col15;
  int n  = (nt << 4) + col15;
  int h = n >> 6, wcol = n & 63;
  const f16* arow0 = w + (size_t)co * K9C + khalf;
  const f16* brow0 = inP + (size_t)b * NP * CC + khalf;
  v8f acc = {};
  for (int ky = 0; ky < 3; ++ky) {
    for (int kx = 0; kx < 3; ++kx) {
      int np = (h + ky) * HP + (wcol + kx);       // always in padded bounds
      const f16* brow = brow0 + (size_t)np * CC;
      const f16* arow = arow0 + (ky * 3 + kx) * CC;
#pragma unroll
      for (int ci = 0; ci < CC; ci += 32)
        acc = wmma16(ld_frag16(arow + ci), ld_frag16(brow + ci), acc);
    }
  }
#pragma unroll
  for (int r = 0; r < 8; ++r) {
    int ch = (ct << 4) + r + ((lane >> 4) << 3);
    int px = (nt << 4) + col15;
    float v = acc[r] + bias[ch];
    v = fmaxf(v, 0.f);
    size_t li = ((size_t)b * CC + ch) * NN + px;
    if (pos)    v += pos[li];
    if (lin32)  lin32[li] = v;
    if (lin16)  lin16[li] = (f16)v;
    if (outT16) outT16[((size_t)b * NN + px) * CC + ch] = (f16)v;
  }
}

// ---- 1x1 conv (q/k/v) as GEMM: out[n][co] = yT16[n][:] . w[co][:] ----------
__global__ void __launch_bounds__(256)
k_lin(const f16* __restrict__ A, const f16* __restrict__ Bw,
      const float* __restrict__ bias, int cols,
      float* __restrict__ rm32, f16* __restrict__ rm16,
      float* __restrict__ cm32, f16* __restrict__ cm16) {
  int wid  = (blockIdx.x * 256 + threadIdx.x) >> 5;
  int lane = threadIdx.x & 31;
  int ctiles = cols >> 4;
  if (wid >= (BB * NN / 16) * ctiles) return;
  int ct = wid % ctiles;
  int rt = wid / ctiles;
  int col15 = lane & 15;
  int khalf = (lane >> 4) << 3;
  const f16* arow = A  + (size_t)((rt << 4) + col15) * CC + khalf;
  const f16* brow = Bw + (size_t)((ct << 4) + col15) * CC + khalf;
  v8f acc = {};
#pragma unroll
  for (int ci = 0; ci < CC; ci += 32)
    acc = wmma16(ld_frag16(arow + ci), ld_frag16(brow + ci), acc);
#pragma unroll
  for (int r = 0; r < 8; ++r) {
    int row = (rt << 4) + r + ((lane >> 4) << 3);
    int ch  = (ct << 4) + col15;
    float v = acc[r] + bias[ch];
    int b = row >> 12, n = row & (NN - 1);
    if (rm32) rm32[(size_t)row * cols + ch] = v;
    if (rm16) rm16[(size_t)row * cols + ch] = (f16)v;
    if (cm32) cm32[((size_t)b * cols + ch) * NN + n] = v;
    if (cm16) cm16[((size_t)b * cols + ch) * NN + n] = (f16)v;
  }
}

// ---- energy = q16 @ k16^T (fp16 WMMA, K=64) --------------------------------
__global__ void __launch_bounds__(256)
k_energy(const f16* __restrict__ qT, const f16* __restrict__ kT,
         f16* __restrict__ attn) {
  int wid  = (blockIdx.x * 256 + threadIdx.x) >> 5;
  int lane = threadIdx.x & 31;
  if (wid >= BB * 256 * 256) return;
  int mt = wid & 255;
  int qt = (wid >> 8) & 255;
  int b  = wid >> 16;
  int col15 = lane & 15;
  int khalf = (lane >> 4) << 3;
  const f16* arow = qT + ((size_t)b * NN + (qt << 4) + col15) * DD + khalf;
  const f16* brow = kT + ((size_t)b * NN + (mt << 4) + col15) * DD + khalf;
  v8f acc = {};
  acc = wmma16(ld_frag16(arow),      ld_frag16(brow),      acc);
  acc = wmma16(ld_frag16(arow + 32), ld_frag16(brow + 32), acc);
  size_t base = (size_t)b * NN * NN;
#pragma unroll
  for (int r = 0; r < 8; ++r) {
    int q = (qt << 4) + r + ((lane >> 4) << 3);
    int m = (mt << 4) + col15;
    attn[base + (size_t)q * NN + m] = (f16)acc[r];
  }
}

// ---- row softmax over N=4096 (in place, fp32 math) -------------------------
__global__ void k_softmax(f16* __restrict__ attn) {
  __shared__ float red[256];
  f16* p = attn + (size_t)blockIdx.x * NN;
  int t = threadIdx.x;
  float mx = -3.0e38f;
  for (int i = t; i < NN; i += 256) mx = fmaxf(mx, (float)p[i]);
  red[t] = mx; __syncthreads();
  for (int s = 128; s > 0; s >>= 1) { if (t < s) red[t] = fmaxf(red[t], red[t + s]); __syncthreads(); }
  float m = red[0]; __syncthreads();
  float sm = 0.f;
  for (int i = t; i < NN; i += 256) sm += __expf((float)p[i] - m);
  red[t] = sm; __syncthreads();
  for (int s = 128; s > 0; s >>= 1) { if (t < s) red[t] += red[t + s]; __syncthreads(); }
  float inv = 1.0f / red[0];
  for (int i = t; i < NN; i += 256) p[i] = (f16)(__expf((float)p[i] - m) * inv);
}

// ---- importance[b][m] = sum_q attn[b][q][m] --------------------------------
__global__ void k_colsum(const f16* __restrict__ attn, float* __restrict__ imp) {
  int g = blockIdx.x * 256 + threadIdx.x;
  if (g >= BB * NN) return;
  int b = g >> 12, m = g & (NN - 1);
  const f16* p = attn + (size_t)b * NN * NN + m;
  float s = 0.f;
  for (int q = 0; q < NN; ++q) s += (float)p[(size_t)q * NN];
  imp[g] = s;
}

// ---- per-batch top-K via bitonic sort of 4096 (val,idx) in LDS -------------
__global__ void __launch_bounds__(1024)
k_topk(const float* __restrict__ imp, int* __restrict__ idx,
       unsigned char* __restrict__ mask) {
  __shared__ float sv[NN];
  __shared__ int   si[NN];
  int b = blockIdx.x, t = threadIdx.x;
  for (int i = t; i < NN; i += 1024) { sv[i] = imp[b * NN + i]; si[i] = i; }
  __syncthreads();
  for (int k2 = 2; k2 <= NN; k2 <<= 1) {
    for (int j = k2 >> 1; j > 0; j >>= 1) {
      for (int i = t; i < NN; i += 1024) {
        int ixj = i ^ j;
        if (ixj > i) {
          bool desc = ((i & k2) == 0);
          float a = sv[i], c = sv[ixj];
          bool sw = desc ? (a < c) : (a > c);
          if (sw) {
            sv[i] = c; sv[ixj] = a;
            int ti = si[i]; si[i] = si[ixj]; si[ixj] = ti;
          }
        }
      }
      __syncthreads();
    }
  }
  for (int i = t; i < NN; i += 1024) mask[b * NN + i] = 0;
  __syncthreads();
  for (int i = t; i < KTOP; i += 1024) { idx[b * 1024 + i] = si[i]; mask[b * NN + si[i]] = 1; }
}

// ---- coarse out16 = v16 @ attn^T (fp16 WMMA, K=4096) -----------------------
// All 8 waves of a block share one 16x4096 fp16 A-tile of v16 (128 KB):
// stage it once via the Tensor Data Mover, then read A-fragments from LDS.
__global__ void __launch_bounds__(256)
k_av(const f16* __restrict__ vmat, const f16* __restrict__ attn,
     f16* __restrict__ out16) {
  extern __shared__ f16 smem[];                  // 16 * NN halfs = 128 KB
  int wid0 = blockIdx.x * 8;
  int ct = (wid0 >> 8) & 15;                     // block-uniform (qt is fastest dim)
  int b  = wid0 >> 12;
  if (threadIdx.x < 32) {                        // wave 0 issues the TDM load
    const f16* gsrc = vmat + ((size_t)b * CC + (ct << 4)) * NN;
    unsigned long long ga = (unsigned long long)(size_t)gsrc;
    unsigned lds_off = (unsigned)(size_t)(void*)smem;
    v4u g0;
    g0[0] = 1u;                                  // count=1 (valid), user mode
    g0[1] = lds_off;                             // D#.lds_addr
    g0[2] = (unsigned)ga;                        // D#.global_addr[31:0]
    g0[3] = (unsigned)((ga >> 32) & 0x01ffffffull) | (2u << 30);  // addr[56:32], type=2
    v8i g1;
    g1[0] = 1 << 16;                             // data_size=1 (2 bytes), wg_mask=0
    g1[1] = (int)(4096u << 16);                  // tensor_dim0[15:0] = 4096
    g1[2] = (int)(16u << 16);                    // tensor_dim0 hi, tensor_dim1 = 16
    g1[3] = (int)(4096u << 16);                  // tensor_dim1 hi, tile_dim0 = 4096
    g1[4] = 16;                                  // tile_dim1 = 16, tile_dim2 = 0
    g1[5] = 4096;                                // tensor_dim0_stride[31:0] = 4096
    g1[6] = 0;
    g1[7] = 0;                                   // no dim1 stride (2D)
    v4i zz; zz[0] = 0; zz[1] = 0; zz[2] = 0; zz[3] = 0;
    v8i z8; z8[0] = 0; z8[1] = 0; z8[2] = 0; z8[3] = 0;
    z8[4] = 0; z8[5] = 0; z8[6] = 0; z8[7] = 0;
    // 6-arg form on this toolchain: (g0, g1, g2, g3, g4, cpol)
    __builtin_amdgcn_tensor_load_to_lds(g0, g1, zz, zz, z8, 0);
    __builtin_amdgcn_s_wait_tensorcnt(0);
  }
  __syncthreads();
  int wid  = wid0 + (threadIdx.x >> 5);
  int lane = threadIdx.x & 31;
  int qt = wid & 255;
  int col15 = lane & 15;
  int khalf = (lane >> 4) << 3;
  const f16* arow = smem + (size_t)col15 * NN + khalf;                        // LDS
  const f16* brow = attn + ((size_t)b * NN + (qt << 4) + col15) * NN + khalf; // global
  v8f acc = {};
  for (int kk = 0; kk < NN; kk += 32) {
    if ((kk & 255) == 0) __builtin_prefetch(brow + kk + 512, 0, 1);
    acc = wmma16(ld_frag16(arow + kk), ld_frag16(brow + kk), acc);
  }
#pragma unroll
  for (int r = 0; r < 8; ++r) {
    int ch = (ct << 4) + r + ((lane >> 4) << 3);
    int q  = (qt << 4) + col15;
    out16[((size_t)b * CC + ch) * NN + q] = (f16)acc[r];
  }
}

// ---- gathers for fp32 refinement -------------------------------------------
__global__ void k_gather_qk(const float* __restrict__ qT32, const float* __restrict__ kT32,
                            const int* __restrict__ idx, float* __restrict__ qg,
                            float* __restrict__ kg) {
  int g = blockIdx.x * 256 + threadIdx.x;
  if (g >= BB * KPAD * DD) return;
  int b = g / (KPAD * DD);
  int r = g - b * (KPAD * DD);
  int kt = r / DD, dd = r - kt * DD;
  float qv = 0.f, kv = 0.f;
  if (kt < KTOP) {
    int id = idx[b * 1024 + kt];
    size_t s = ((size_t)b * NN + id) * DD + dd;
    qv = qT32[s]; kv = kT32[s];
  }
  qg[g] = qv; kg[g] = kv;
}

__global__ void k_gather_v(const float* __restrict__ v32, const int* __restrict__ idx,
                           float* __restrict__ vg) {
  int g = blockIdx.x * 256 + threadIdx.x;
  if (g >= BB * CC * KPAD) return;
  int b = g / (CC * KPAD);
  int r = g - b * (CC * KPAD);
  int c = r / KPAD, kt = r - c * KPAD;
  vg[g] = (kt < KTOP) ? v32[((size_t)b * CC + c) * NN + idx[b * 1024 + kt]] : 0.f;
}

// ---- refinement energy: fp32 WMMA 16x16x4 ----------------------------------
__global__ void __launch_bounds__(256)
k_eref(const float* __restrict__ qg, const float* __restrict__ kg,
       float* __restrict__ eref) {
  int wid  = (blockIdx.x * 256 + threadIdx.x) >> 5;
  int lane = threadIdx.x & 31;
  if (wid >= BB * 52 * 52) return;
  int jt = wid % 52;
  int t2 = wid / 52;
  int it = t2 % 52;
  int b  = t2 / 52;
  int col15 = lane & 15;
  int kq = (lane >> 4) << 1;
  const float* arow = qg + ((size_t)b * KPAD + (it << 4) + col15) * DD + kq;
  const float* brow = kg + ((size_t)b * KPAD + (jt << 4) + col15) * DD + kq;
  v8f acc = {};
#pragma unroll
  for (int kk = 0; kk < DD; kk += 4)
    acc = wmma4(ld2f(arow + kk), ld2f(brow + kk), acc);
#pragma unroll
  for (int r = 0; r < 8; ++r) {
    int i = (it << 4) + r + ((lane >> 4) << 3);
    int j = (jt << 4) + col15;
    eref[((size_t)b * KPAD + i) * KPAD + j] = acc[r];
  }
}

__global__ void k_softmax_ref(float* __restrict__ eref) {
  __shared__ float red[256];
  int b = blockIdx.x / KPAD;
  int i = blockIdx.x - b * KPAD;
  float* row = eref + ((size_t)b * KPAD + i) * KPAD;
  int t = threadIdx.x;
  if (i >= KTOP) {
    for (int j = t; j < KPAD; j += 256) row[j] = 0.f;
    return;
  }
  float mx = -3.0e38f;
  for (int j = t; j < KTOP; j += 256) mx = fmaxf(mx, row[j]);
  red[t] = mx; __syncthreads();
  for (int s = 128; s > 0; s >>= 1) { if (t < s) red[t] = fmaxf(red[t], red[t + s]); __syncthreads(); }
  float m = red[0]; __syncthreads();
  float sm = 0.f;
  for (int j = t; j < KTOP; j += 256) sm += __expf(row[j] - m);
  red[t] = sm; __syncthreads();
  for (int s = 128; s > 0; s >>= 1) { if (t < s) red[t] += red[t + s]; __syncthreads(); }
  float inv = 1.0f / red[0];
  for (int j = t; j < KTOP; j += 256) row[j] = __expf(row[j] - m) * inv;
  for (int j = KTOP + t; j < KPAD; j += 256) row[j] = 0.f;
}

// ---- out_topk = vk @ aref^T (fp32 WMMA) ------------------------------------
__global__ void __launch_bounds__(256)
k_outtop(const float* __restrict__ vg, const float* __restrict__ eref,
         float* __restrict__ ot) {
  int wid  = (blockIdx.x * 256 + threadIdx.x) >> 5;
  int lane = threadIdx.x & 31;
  if (wid >= BB * 16 * 52) return;
  int qt = wid % 52;
  int t2 = wid / 52;
  int ct = t2 % 16;
  int b  = t2 / 16;
  int col15 = lane & 15;
  int kq = (lane >> 4) << 1;
  const float* arow = vg   + ((size_t)b * CC   + (ct << 4) + col15) * KPAD + kq;
  const float* brow = eref + ((size_t)b * KPAD + (qt << 4) + col15) * KPAD + kq;
  v8f acc = {};
  for (int kk = 0; kk < KPAD; kk += 4)
    acc = wmma4(ld2f(arow + kk), ld2f(brow + kk), acc);
#pragma unroll
  for (int r = 0; r < 8; ++r) {
    int ch = (ct << 4) + r + ((lane >> 4) << 3);
    int q  = (qt << 4) + col15;
    ot[((size_t)b * CC + ch) * KPAD + q] = acc[r];
  }
}

// ---- assemble fp32 path, scatter refinement, combine -----------------------
__global__ void k_prep32(const f16* __restrict__ o16, float* __restrict__ o32) {
  int g = blockIdx.x * 256 + threadIdx.x;
  if (g < BB * CC * NN) o32[g] = (float)o16[g];
}

__global__ void k_scatter(const float* __restrict__ ot, const int* __restrict__ idx,
                          float* __restrict__ o32) {
  int g = blockIdx.x * 256 + threadIdx.x;
  if (g >= BB * CC * KTOP) return;
  int b = g / (CC * KTOP);
  int r = g - b * (CC * KTOP);
  int c = r / KTOP, kt = r - c * KTOP;
  o32[((size_t)b * CC + c) * NN + idx[b * 1024 + kt]] =
      ot[((size_t)b * CC + c) * KPAD + kt];
}

// combine into padded (66x66) fp16 layouts for the final convs; halo = 0
__global__ void k_combine(const f16* __restrict__ o16, const float* __restrict__ o32,
                          const float* __restrict__ y32, const float* __restrict__ gamma,
                          f16* __restrict__ oAT, f16* __restrict__ oBT) {
  int g = blockIdx.x * 256 + threadIdx.x;
  if (g >= BB * NP * CC) return;
  int b = g / (NP * CC);
  int r = g - b * (NP * CC);
  int np = r / CC;
  int c  = r - np * CC;
  int hh = np / HP, ww = np - hh * HP;
  int h = hh - 1, w = ww - 1;
  f16 av = (f16)0.f, bv = (f16)0.f;
  if (h >= 0 && h < HH && w >= 0 && w < WW) {
    size_t src = ((size_t)b * CC + c) * NN + (h << 6) + w;
    float gm = gamma[0];
    float y = y32[src];
    av = (f16)(gm * (float)o16[src] + (float)(f16)y);  // gamma*out16 + x.f16 promote
    bv = (f16)(gm * o32[src] + y);
  }
  oAT[g] = av;
  oBT[g] = bv;
}

// ---- host orchestration -----------------------------------------------------
extern "C" void kernel_launch(void* const* d_in, const int* in_sizes, int n_in,
                              void* d_out, int out_size, void* d_ws, size_t ws_size,
                              hipStream_t stream) {
  (void)in_sizes; (void)n_in; (void)out_size; (void)ws_size;
  const float* x     = (const float*)d_in[0];
  const float* pos   = (const float*)d_in[1];
  const float* pre_w = (const float*)d_in[2];
  const float* pre_s = (const float*)d_in[3];
  const float* pre_b = (const float*)d_in[4];
  const float* pre_m = (const float*)d_in[5];
  const float* pre_v = (const float*)d_in[6];
  const float* q_w   = (const float*)d_in[7];
  const float* q_b   = (const float*)d_in[8];
  const float* k_w   = (const float*)d_in[9];
  const float* k_b   = (const float*)d_in[10];
  const float* v_w   = (const float*)d_in[11];
  const float* v_b   = (const float*)d_in[12];
  const float* fin_w = (const float*)d_in[13];
  const float* fin_s = (const float*)d_in[14];
  const float* fin_b = (const float*)d_in[15];
  const float* fin_m = (const float*)d_in[16];
  const float* fin_v = (const float*)d_in[17];
  const float* gamma = (const float*)d_in[18];

  char* wsb = (char*)d_ws;
  size_t off = 0;
  auto take = [&](size_t bytes) -> void* {
    void* p = wsb + off;
    off = (off + bytes + 255) & ~(size_t)255;
    return p;
  };
  f16*   wpre16 = (f16*)  take((size_t)CC * K9C * 2);
  float* bpre   = (float*)take(CC * 4);
  f16*   wfin16 = (f16*)  take((size_t)CC * K9C * 2);
  float* bfin   = (float*)take(CC * 4);
  f16*   wq16   = (f16*)  take((size_t)DD * CC * 2);
  f16*   wk16   = (f16*)  take((size_t)DD * CC * 2);
  f16*   wv16   = (f16*)  take((size_t)CC * CC * 2);
  f16*   xinP   = (f16*)  take((size_t)BB * NP * CC * 2);
  float* y32    = (float*)take((size_t)BB * CC * NN * 4);
  f16*   yT16   = (f16*)  take((size_t)BB * NN * CC * 2);
  float* qT32   = (float*)take((size_t)BB * NN * DD * 4);
  f16*   qT16   = (f16*)  take((size_t)BB * NN * DD * 2);
  float* kT32   = (float*)take((size_t)BB * NN * DD * 4);
  f16*   kT16   = (f16*)  take((size_t)BB * NN * DD * 2);
  float* v32    = (float*)take((size_t)BB * CC * NN * 4);
  f16*   v16b   = (f16*)  take((size_t)BB * CC * NN * 2);
  f16*   attn   = (f16*)  take((size_t)BB * NN * NN * 2);
  float* imp    = (float*)take((size_t)BB * NN * 4);
  int*   idx    = (int*)  take((size_t)BB * 1024 * 4);
  float* qg     = (float*)take((size_t)BB * KPAD * DD * 4);
  float* kg     = (float*)take((size_t)BB * KPAD * DD * 4);
  float* vg     = (float*)take((size_t)BB * CC * KPAD * 4);
  float* eref   = (float*)take((size_t)BB * KPAD * KPAD * 4);
  float* ot     = (float*)take((size_t)BB * CC * KPAD * 4);
  f16*   out16b = (f16*)  take((size_t)BB * CC * NN * 2);
  float* out32b = (float*)take((size_t)BB * CC * NN * 4);
  f16*   oAT    = (f16*)  take((size_t)BB * NP * CC * 2);
  f16*   oBT    = (f16*)  take((size_t)BB * NP * CC * 2);

  f16*   out16p = (f16*)d_out;
  float* out32p = (float*)((char*)d_out + (size_t)BB * CC * NN * 2);
  unsigned char* maskp =
      (unsigned char*)((char*)d_out + (size_t)BB * CC * NN * 2 + (size_t)BB * CC * NN * 4);

  dim3 blk(256);
  // weight prep + padded input transpose
  k_fold3<<<(CC * CC * 9 + 255) / 256, blk, 0, stream>>>(pre_w, pre_s, pre_b, pre_m, pre_v, wpre16, bpre);
  k_fold3<<<(CC * CC * 9 + 255) / 256, blk, 0, stream>>>(fin_w, fin_s, fin_b, fin_m, fin_v, wfin16, bfin);
  k_pack1<<<(DD * CC + 255) / 256, blk, 0, stream>>>(q_w, wq16, DD * CC);
  k_pack1<<<(DD * CC + 255) / 256, blk, 0, stream>>>(k_w, wk16, DD * CC);
  k_pack1<<<(CC * CC + 255) / 256, blk, 0, stream>>>(v_w, wv16, CC * CC);
  k_tin_pad<<<(BB * NP * CC + 255) / 256, blk, 0, stream>>>(x, xinP);
  // pre conv -> y (x after BN/ReLU + pos), fp32 lin + fp16 transposed
  k_conv3<<<(BB * (NN / 16) * (CC / 16)) * 32 / 256, blk, 0, stream>>>(
      xinP, wpre16, bpre, pos, y32, nullptr, yT16);
  // q/k/v 1x1 convs
  k_lin<<<((BB * NN / 16) * (DD / 16)) * 32 / 256, blk, 0, stream>>>(
      yT16, wq16, q_b, DD, qT32, qT16, nullptr, nullptr);
  k_lin<<<((BB * NN / 16) * (DD / 16)) * 32 / 256, blk, 0, stream>>>(
      yT16, wk16, k_b, DD, kT32, kT16, nullptr, nullptr);
  k_lin<<<((BB * NN / 16) * (CC / 16)) * 32 / 256, blk, 0, stream>>>(
      yT16, wv16, v_b, CC, nullptr, nullptr, v32, v16b);
  // coarse attention
  k_energy<<<(BB * 256 * 256) * 32 / 256, blk, 0, stream>>>(qT16, kT16, attn);
  k_softmax<<<BB * NN, blk, 0, stream>>>(attn);
  k_colsum<<<(BB * NN + 255) / 256, blk, 0, stream>>>(attn, imp);
  k_topk<<<BB, dim3(1024), 0, stream>>>(imp, idx, maskp);
  // v @ attn^T with TDM-staged A tile (128 KB dynamic LDS per workgroup)
  k_av<<<(BB * 16 * 256) * 32 / 256, blk, 16 * NN * sizeof(f16), stream>>>(v16b, attn, out16b);
  // fp32 refinement on top-k
  k_gather_qk<<<(BB * KPAD * DD + 255) / 256, blk, 0, stream>>>(qT32, kT32, idx, qg, kg);
  k_gather_v<<<(BB * CC * KPAD + 255) / 256, blk, 0, stream>>>(v32, idx, vg);
  k_eref<<<(BB * 52 * 52) * 32 / 256, blk, 0, stream>>>(qg, kg, eref);
  k_softmax_ref<<<BB * KPAD, blk, 0, stream>>>(eref);
  k_outtop<<<(BB * 16 * 52) * 32 / 256, blk, 0, stream>>>(vg, eref, ot);
  k_prep32<<<(BB * CC * NN + 255) / 256, blk, 0, stream>>>(out16b, out32b);
  k_scatter<<<(BB * CC * KTOP + 255) / 256, blk, 0, stream>>>(ot, idx, out32b);
  // combine (padded halo layouts) + final convs
  k_combine<<<(BB * NP * CC + 255) / 256, blk, 0, stream>>>(out16b, out32b, y32, gamma, oAT, oBT);
  k_conv3<<<(BB * (NN / 16) * (CC / 16)) * 32 / 256, blk, 0, stream>>>(
      oAT, wfin16, bfin, nullptr, nullptr, out16p, nullptr);
  k_conv3<<<(BB * (NN / 16) * (CC / 16)) * 32 / 256, blk, 0, stream>>>(
      oBT, wfin16, bfin, nullptr, out32p, nullptr, nullptr);
}
